// ModulatedConv2d_79465484910975
// MI455X (gfx1250) — matrix operand order repaired and checked by reading, not compiled, not run on hardware
//
#include <hip/hip_runtime.h>

typedef unsigned short u16;
typedef __attribute__((ext_vector_type(8)))  __bf16 v8bf;
typedef __attribute__((ext_vector_type(16))) __bf16 v16bf;
typedef __attribute__((ext_vector_type(8)))  float  v8f;

// ---- problem constants ----
// B=4, IC=512, OC=512, K=3, D=512, H=W=128
// FC_WLR = 1/sqrt(512*512) = 1/512 ; W_LRMUL = 1/sqrt(512*512*9) = 1/1536

__device__ __forceinline__ u16 f2bf(float f) {           // RNE f32->bf16
  unsigned int u = __float_as_uint(f);
  u += 0x7fffu + ((u >> 16) & 1u);
  return (u16)(u >> 16);
}

// s[b,i] = lrelu(y @ (dw*FC_WLR)^T + db + 1, 0.2) * sqrt(2)        (2048 outputs)
__global__ void k_fc(const float* __restrict__ y, const float* __restrict__ dw,
                     const float* __restrict__ db, float* __restrict__ s) {
  int t = blockIdx.x * 256 + threadIdx.x;   // b*512 + i
  int b = t >> 9, i = t & 511;
  float acc = 0.f;
  for (int d = 0; d < 512; ++d) acc += y[b * 512 + d] * dw[i * 512 + d];
  float v = acc * (1.0f / 512.0f) + db[i] + 1.0f;
  v = (v > 0.f ? v : 0.2f * v) * 1.41421356237309515f;
  s[t] = v;
}

// t[o,i] = sum_{kh,kw} w[o,i,kh,kw]^2                              (262144 outputs)
__global__ void k_wsq(const float* __restrict__ w, float* __restrict__ t) {
  int id = blockIdx.x * 256 + threadIdx.x;  // o*512 + i
  const float* p = w + (size_t)id * 9;
  float acc = 0.f;
#pragma unroll
  for (int k = 0; k < 9; ++k) { float v = p[k]; acc += v * v; }
  t[id] = acc;
}

// d[b,o] = rsqrt(mean_{kh,kw,i}(w^2 s^2) + 1e-8)                    (2048 outputs)
__global__ void k_demod(const float* __restrict__ t, const float* __restrict__ s,
                        float* __restrict__ dc) {
  int id = blockIdx.x * 256 + threadIdx.x;  // b*512 + o
  int b = id >> 9, o = id & 511;
  float acc = 0.f;
  for (int i = 0; i < 512; ++i) { float sv = s[b * 512 + i]; acc += t[o * 512 + i] * sv * sv; }
  dc[id] = rsqrtf(acc * (1.0f / 4608.0f) + 1e-8f);
}

// weff[kk][n][m] = bf16( w[o,i,kh,kw] * s[b,i] * d[b,o] * W_LRMUL )
// n = b*128 + o/4 ; m = (o%4)*512 + i  (the reference's reshape mapping)
__global__ void k_weff(const float* __restrict__ w, const float* __restrict__ s,
                       const float* __restrict__ dc, u16* __restrict__ wf) {
  int id = blockIdx.x * 256 + threadIdx.x;  // n*2048 + m (1M threads)
  int n = id >> 11, m = id & 2047;
  int b = n >> 7;
  int o = ((n & 127) << 2) | (m >> 9);
  int i = m & 511;
  float scale = s[b * 512 + i] * dc[b * 512 + o] * (1.0f / 1536.0f);
  const float* wp = w + ((size_t)o * 512 + i) * 9;
#pragma unroll
  for (int kk = 0; kk < 9; ++kk)
    wf[(size_t)kk * (512 * 2048) + id] = f2bf(wp[kk] * scale);
}

// zero-fill xt halo buffer (130*130*2048 bf16 == 4,326,400 uint4)
__global__ void k_zero(uint4* __restrict__ p) {
  p[(size_t)blockIdx.x * 256 + threadIdx.x] = make_uint4(0u, 0u, 0u, 0u);
}

// xt[(y+1)*130 + (x+1)][m] = bf16(x[m][y][x]) — tiled 32x32 transpose via LDS
__global__ void k_xt(const float* __restrict__ x, u16* __restrict__ xt) {
  __shared__ u16 lds[32][33];
  int m0 = blockIdx.x * 32;   // 64 channel tiles
  int x0 = blockIdx.y * 32;   // 4 column tiles
  int y  = blockIdx.z;        // 128 rows
  int t  = threadIdx.x;
  int c = t & 31, r = t >> 5;
#pragma unroll
  for (int rr = 0; rr < 4; ++rr) {
    int row = r + rr * 8;
    lds[row][c] = f2bf(x[(size_t)(m0 + row) * 16384 + y * 128 + (x0 + c)]);
  }
  __syncthreads();
  int c2 = t & 7;        // 4-channel sub-chunk
  int rx = t >> 3;       // column within tile
  u16 v0 = lds[c2 * 4 + 0][rx], v1 = lds[c2 * 4 + 1][rx];
  u16 v2 = lds[c2 * 4 + 2][rx], v3 = lds[c2 * 4 + 3][rx];
  size_t dst = ((size_t)(y + 1) * 130 + (x0 + rx + 1)) * 2048 + m0 + c2 * 4;
  uint2 pk;
  pk.x = (unsigned)v0 | ((unsigned)v1 << 16);
  pk.y = (unsigned)v2 | ((unsigned)v3 << 16);
  *reinterpret_cast<uint2*>(xt + dst) = pk;
}

// Implicit-GEMM conv: out[n, y*128+x] = sum_{kk,m} weff[kk][n][m] * xt[y+kh][x+kw][m]
// Block: 256 n x 128 px (one image row). 8 waves (4 n x 2 px), wave tile 64x64
// => 4x4 f32 accumulators, 16 WMMA per K=32 step vs 16 b128 loads (32 FLOP/B).
__global__ __launch_bounds__(256) void k_conv(const u16* __restrict__ Wf,
                                              const u16* __restrict__ Xt,
                                              float* __restrict__ out) {
  const int y    = blockIdx.x;          // image row
  const int nb   = blockIdx.y << 8;     // n tile base (256 rows)
  const int lane = threadIdx.x & 31;
  const int wave = threadIdx.x >> 5;
  const int wn   = wave >> 1;           // 0..3 -> n offset 64*wn
  const int wp   = wave & 1;            // 0..1 -> pixel offset 64*wp
  const int half = lane >> 4;           // K-half select per WMMA layout
  const int l15  = lane & 15;           // A row / B column / C column

  const v8f vzero = {0.f, 0.f, 0.f, 0.f, 0.f, 0.f, 0.f, 0.f};
  v8f acc[4][4];
#pragma unroll
  for (int a = 0; a < 4; ++a)
#pragma unroll
    for (int b2 = 0; b2 < 4; ++b2) acc[a][b2] = vzero;

  // A rows: lane l15 holds row n; K chunks at m0 + half*8 and m0 + 16 + half*8
  const u16* arow[4];
#pragma unroll
  for (int a = 0; a < 4; ++a)
    arow[a] = Wf + ((size_t)(nb + wn * 64 + a * 16 + l15)) * 2048 + half * 8;
  const int xc0 = wp * 64 + l15;

  for (int kk = 0; kk < 9; ++kk) {
    const int kh = kk / 3, kw = kk % 3;
    const size_t aoff = (size_t)kk * (512 * 2048);
    // B columns: lane l15 holds pixel column; K chunks at m0 + half*8 and m0 + 16 + half*8
    const u16* brow[4];
#pragma unroll
    for (int b2 = 0; b2 < 4; ++b2)
      brow[b2] = Xt + ((size_t)(y + kh) * 130 + (xc0 + b2 * 16 + kw)) * 2048 + half * 8;

    __builtin_prefetch(brow[0], 0, 0);
    __builtin_prefetch(arow[0] + aoff, 0, 0);

#pragma unroll 2
    for (int m0 = 0; m0 < 2048; m0 += 32) {
      union { v16bf v; v8bf h[2]; } bu[4], au;
#pragma unroll
      for (int b2 = 0; b2 < 4; ++b2) {
        bu[b2].h[0] = *reinterpret_cast<const v8bf*>(brow[b2] + m0);
        bu[b2].h[1] = *reinterpret_cast<const v8bf*>(brow[b2] + m0 + 16);
      }
#pragma unroll
      for (int a = 0; a < 4; ++a) {
        const u16* ap = arow[a] + aoff + m0;
        au.h[0] = *reinterpret_cast<const v8bf*>(ap);
        au.h[1] = *reinterpret_cast<const v8bf*>(ap + 16);
#pragma unroll
        for (int b2 = 0; b2 < 4; ++b2)
          acc[a][b2] = __builtin_amdgcn_wmma_f32_16x16x32_bf16(
              false, au.v, false, bu[b2].v, (short)0, acc[a][b2], false, false);
      }
    }
  }

  // C/D layout: lane -> column N=l15, VGPR j -> row M = j + half*8
#pragma unroll
  for (int a = 0; a < 4; ++a) {
#pragma unroll
    for (int j = 0; j < 8; ++j) {
      const int n = nb + wn * 64 + a * 16 + half * 8 + j;
      float* op = out + (size_t)n * 16384 + (size_t)y * 128 + wp * 64 + l15;
#pragma unroll
      for (int b2 = 0; b2 < 4; ++b2) op[b2 * 16] = acc[a][b2][j];
    }
  }
}

extern "C" void kernel_launch(void* const* d_in, const int* in_sizes, int n_in,
                              void* d_out, int out_size, void* d_ws, size_t ws_size,
                              hipStream_t stream) {
  (void)in_sizes; (void)n_in; (void)out_size; (void)ws_size;
  const float* x  = (const float*)d_in[0];
  const float* yv = (const float*)d_in[1];
  const float* w  = (const float*)d_in[2];
  const float* dw = (const float*)d_in[3];
  const float* db = (const float*)d_in[4];
  float* out = (float*)d_out;

  // workspace layout (~85 MB total)
  float* s  = (float*)d_ws;                 // 2048 f32
  float* dc = s + 2048;                     // 2048 f32
  float* tb = dc + 2048;                    // 262144 f32
  u16*   wf = (u16*)(tb + 512 * 512);       // 9*512*2048 bf16 (18.9 MB)
  u16*   xt = wf + (size_t)9 * 512 * 2048;  // 130*130*2048 bf16 (69.2 MB)

  k_fc   <<<8,    256, 0, stream>>>(yv, dw, db, s);
  k_wsq  <<<1024, 256, 0, stream>>>(w, tb);
  k_demod<<<8,    256, 0, stream>>>(tb, s, dc);
  k_weff <<<4096, 256, 0, stream>>>(w, s, dc, wf);
  k_zero <<<16900, 256, 0, stream>>>((uint4*)xt);
  k_xt   <<<dim3(64, 4, 128), 256, 0, stream>>>(x, xt);
  k_conv <<<dim3(128, 2),     256, 0, stream>>>(wf, xt, out);
}
